// QHNet_37847251812863
// MI455X (gfx1250) — matrix operand order repaired and checked by reading, not compiled, not run on hardware
//
#include <hip/hip_runtime.h>
#include <hip/hip_bf16.h>
#include <math.h>

// ----------------------------- problem constants -----------------------------
#define EDG   17664      // E = M*A*(A-1)   (multiple of 64)
#define NND   768        // N = M*A         (multiple of 64)
#define HSZ   128        // HS
#define HBSZ  32         // HBS
#define ORB   14
#define SDIM  25         // sum(2l+1), l=0..4
#define NPWC  56
#define RADW  1280       // 5*2*HS
#define NPE   23         // edges per dst node (A-1); edges are sorted by dst
#define SQ4PI 3.5449077018110318f

typedef __attribute__((ext_vector_type(16))) _Float16     v16h;
typedef __attribute__((ext_vector_type(8)))  float        v8f;
typedef __attribute__((ext_vector_type(4)))  unsigned int v4u;
typedef __attribute__((ext_vector_type(8)))  int          v8i;
typedef __attribute__((ext_vector_type(4)))  int          v4i;

// l for each of the 25 (l,m) slots
__constant__ int c_SL[SDIM] = {0,1,1,1,2,2,2,2,2,3,3,3,3,3,3,3,4,4,4,4,4,4,4,4,4};
// expansion term tables, PATHS visit order (19 (path,l) terms)
#define NTERM 19
__constant__ int c_TL [NTERM] = {0,1,2,1,0,1,2,1,2,3,2,1,2,3,0,1,2,3,4};
__constant__ int c_TL1[NTERM] = {0,0,0,1,1,1,1,1,1,1,2,2,2,2,2,2,2,2,2};
__constant__ int c_TL2[NTERM] = {0,1,2,0,1,1,1,2,2,2,0,1,1,1,2,2,2,2,2};
__constant__ int c_TOFF[NTERM]= {0,1,10,35,44,53,80,125,170,245,350,375,420,495,600,625,700,825,1000};
__constant__ int c_TPW[NTERM] = {0,9,15,18,24,28,32,36,38,40,42,45,47,49,51,52,53,54,55};
__constant__ int c_LOFF[5] = {0,1,4,9,16};
__constant__ int c_MUL[3]  = {3,2,1};
__constant__ int c_ROFF[3] = {0,3,9};

// ------------------------------------------------------------------------------
// Generic WMMA GEMM:  C[M,Nn] = act(A[M,K] @ B[K,Nn] + bias)  (+= optional)
//   * M must be a multiple of 64 (true for every call site) -> no row guards.
//   * A/C row addressing: row r -> base + (r/inner)*outer + (r%inner)*ld, with
//     the division done by magic-multiply (exact for inner<=9, r<2^31).
//   * B tile (32k x 64n, f32) is DMA'd into LDS by the Tensor Data Mover: one
//     wave issues tensor_load_to_lds, s_wait_tensorcnt 0, then the workgroup
//     barrier publishes it.
//   * Block tile 64x64: 4 waves in 2x2, each wave owns a 32x32 quadrant ->
//     2 A-fragments x 2 B-fragments -> 4 v_wmma_f32_16x16x32_f16 per K-step.
// flags: bit0 = SiLU epilogue, bit1 = accumulate into C.
// ------------------------------------------------------------------------------
__global__ __launch_bounds__(128)
void gemm_f16_wmma(const float* __restrict__ A, int ai, unsigned amag, long ao, int al,
                   const float* __restrict__ B, int ldb,
                   const float* __restrict__ bias,
                   float* __restrict__ C, int ci, unsigned cmag, long co, int cl,
                   int M, int Nn, int K, int flags)
{
  __shared__ _Float16 Ah[64][34];   // [row][k]   (f16 staged)
  __shared__ float    Bs[32][64];   // [k][n]     (TDM tile, row-major)
  const int tid  = threadIdx.x;
  const int lane = tid & 31;
  const int wv   = tid >> 5;
  const int wrow = wv >> 1, wcol = wv & 1;
  const int lo   = lane & 15, hi = lane >> 4;
  const long tm  = (long)blockIdx.y * 64;
  const int  tn  = blockIdx.x * 64;

  v8f acc00 = {}, acc01 = {}, acc10 = {}, acc11 = {};
  for (int k0 = 0; k0 < K; k0 += 32) {
    // ---- issue TDM load of the B tile (wave 0 only; EXEC-independent) ----
    if (wv == 0) {
      unsigned lds = (unsigned)(unsigned long long)(&Bs[0][0]);
      unsigned long long ga =
          (unsigned long long)(const void*)(B + (long)k0 * ldb + tn);
      int dim0 = Nn - tn;       // remaining columns (OOB reads -> zero fill)
      int dim1 = K - k0;        // remaining k rows
      v4u g0;
      g0.x = 1u;                                    // count=1, user descriptor
      g0.y = lds;                                   // lds_addr (bytes)
      g0.z = (unsigned)(ga & 0xffffffffu);          // global_addr[31:0]
      g0.w = (unsigned)((ga >> 32) & 0x01ffffffu)   // global_addr[56:32]
           | (2u << 30);                            // type = 2 ("image")
      v8i g1;
      g1[0] = 0x00020000;                           // data_size = 4B
      g1[1] = (dim0 & 0xffff) << 16;                // tensor_dim0[15:0]
      g1[2] = ((dim0 >> 16) & 0xffff) | ((dim1 & 0xffff) << 16);
      g1[3] = ((dim1 >> 16) & 0xffff) | (64 << 16); // tile_dim0 = 64
      g1[4] = 32;                                   // tile_dim1 = 32
      g1[5] = ldb;                                  // tensor_dim0_stride[31:0]
      g1[6] = 0;
      g1[7] = 0;
      v4i z4 = {0, 0, 0, 0};
      v8i z8 = {0, 0, 0, 0, 0, 0, 0, 0};
      __builtin_amdgcn_tensor_load_to_lds(g0, g1, z4, z4, z8, 0);
    }
    // ---- stage A tile as f16: unguarded (M % 64 == 0), load-all then store ----
    float av[16];
#pragma unroll
    for (int it = 0; it < 16; ++it) {
      int e = tid + it * 128;           // 64 rows x 32 cols = 2048 elements
      int r = e >> 5, c = e & 31;
      unsigned gm = (unsigned)tm + (unsigned)r;
      unsigned q  = (ai == 1) ? gm
                   : (unsigned)(((unsigned long long)gm * amag) >> 40);
      unsigned rm = gm - q * (unsigned)ai;
      av[it] = A[(long)q * ao + (long)rm * al + (k0 + c)];
    }
#pragma unroll
    for (int it = 0; it < 16; ++it) {
      int e = tid + it * 128;
      Ah[e >> 5][e & 31] = (_Float16)av[it];
    }
    if (k0 + 32 < K) {                  // prefetch next A k-block (this thread's row)
      unsigned gm = (unsigned)tm + (unsigned)(tid >> 5);
      unsigned q  = (ai == 1) ? gm
                   : (unsigned)(((unsigned long long)gm * amag) >> 40);
      unsigned rm = gm - q * (unsigned)ai;
      __builtin_prefetch(A + (long)q * ao + (long)rm * al + (k0 + 32), 0, 1);
    }
    if (wv == 0) __builtin_amdgcn_s_wait_tensorcnt(0);
    __syncthreads();

    // ---- fragments in contiguous k = 16*hi + s order (A and B alike); each fragment reused by 2 WMMAs ----
    v16h af0, af1, bf0, bf1;
    const int ar0 = wrow * 32 + lo;
    const int ar1 = ar0 + 16;
    const int nc0 = wcol * 32 + lo;
#pragma unroll
    for (int i = 0; i < 8; ++i) {
      int ka = hi * 16 + 2 * i;
      int kb = hi * 16 + 2 * i;
      af0[2 * i]     = Ah[ar0][ka];
      af0[2 * i + 1] = Ah[ar0][ka + 1];
      af1[2 * i]     = Ah[ar1][ka];
      af1[2 * i + 1] = Ah[ar1][ka + 1];
      bf0[2 * i]     = (_Float16)Bs[kb][nc0];
      bf0[2 * i + 1] = (_Float16)Bs[kb + 1][nc0];
      bf1[2 * i]     = (_Float16)Bs[kb][nc0 + 16];
      bf1[2 * i + 1] = (_Float16)Bs[kb + 1][nc0 + 16];
    }
    acc00 = __builtin_amdgcn_wmma_f32_16x16x32_f16(
        false, af0, false, bf0, (short)0, acc00, false, false);
    acc01 = __builtin_amdgcn_wmma_f32_16x16x32_f16(
        false, af0, false, bf1, (short)0, acc01, false, false);
    acc10 = __builtin_amdgcn_wmma_f32_16x16x32_f16(
        false, af1, false, bf0, (short)0, acc10, false, false);
    acc11 = __builtin_amdgcn_wmma_f32_16x16x32_f16(
        false, af1, false, bf1, (short)0, acc11, false, false);
    __syncthreads();
  }

  // ---- epilogue: D layout vgpr r -> M = r + hi*8, N = lo ----
  const long gn0 = tn + wcol * 32 + lo;
  const long gn1 = gn0 + 16;
  const bool do_silu = (flags & 1) != 0;
  const bool do_acc  = (flags & 2) != 0;
  const float bv0 = (bias && gn0 < Nn) ? bias[gn0] : 0.f;
  const float bv1 = (bias && gn1 < Nn) ? bias[gn1] : 0.f;
#pragma unroll
  for (int rb = 0; rb < 2; ++rb) {
    const v8f a0 = rb ? acc10 : acc00;
    const v8f a1 = rb ? acc11 : acc01;
#pragma unroll
    for (int r = 0; r < 8; ++r) {
      unsigned gm = (unsigned)tm + (unsigned)(wrow * 32 + rb * 16 + hi * 8 + r);
      unsigned q  = (ci == 1) ? gm
                   : (unsigned)(((unsigned long long)gm * cmag) >> 40);
      unsigned rm = gm - q * (unsigned)ci;
      float* rowp = C + (long)q * co + (long)rm * cl;
      if (gn0 < Nn) {
        float v = a0[r] + bv0;
        if (do_silu) v = v / (1.f + __expf(-v));
        if (do_acc)  v += rowp[gn0];
        rowp[gn0] = v;
      }
      if (gn1 < Nn) {
        float v = a1[r] + bv1;
        if (do_silu) v = v / (1.f + __expf(-v));
        if (do_acc)  v += rowp[gn1];
        rowp[gn1] = v;
      }
    }
  }
}

// ------------------------------- small kernels --------------------------------
__global__ void geom_kernel(const float* __restrict__ pos, const int* __restrict__ eidx,
                            float* __restrict__ Y, float* __restrict__ rbuf)
{
  int e = blockIdx.x * blockDim.x + threadIdx.x;
  if (e >= EDG) return;
  int d = eidx[e], s = eidx[EDG + e];
  float vx = pos[d*3+0]-pos[s*3+0], vy = pos[d*3+1]-pos[s*3+1], vz = pos[d*3+2]-pos[s*3+2];
  float r = sqrtf(vx*vx + vy*vy + vz*vz);
  rbuf[e] = r;
  float inv = 1.f / r;
  // component permutation [:, [1,2,0]]
  float x = vy*inv, y = vz*inv, z = vx*inv;
  float x2 = x*x, y2 = y*y, z2 = z*z;
  float* Ye = Y + (long)e * SDIM;
  const float C = SQ4PI;
  Ye[0]  = C*0.28209479177387814f;
  Ye[1]  = C*0.4886025119029199f*y;
  Ye[2]  = C*0.4886025119029199f*z;
  Ye[3]  = C*0.4886025119029199f*x;
  Ye[4]  = C*1.0925484305920792f*x*y;
  Ye[5]  = C*1.0925484305920792f*y*z;
  Ye[6]  = C*0.31539156525252005f*(3.f*z2 - 1.f);
  Ye[7]  = C*1.0925484305920792f*x*z;
  Ye[8]  = C*0.5462742152960396f*(x2 - y2);
  Ye[9]  = C*0.5900435899266435f*y*(3.f*x2 - y2);
  Ye[10] = C*2.890611442640554f*x*y*z;
  Ye[11] = C*0.4570457994644658f*y*(5.f*z2 - 1.f);
  Ye[12] = C*0.3731763325901154f*z*(5.f*z2 - 3.f);
  Ye[13] = C*0.4570457994644658f*x*(5.f*z2 - 1.f);
  Ye[14] = C*1.445305721320277f*z*(x2 - y2);
  Ye[15] = C*0.5900435899266435f*x*(x2 - 3.f*y2);
  Ye[16] = C*2.5033429417967046f*x*y*(x2 - y2);
  Ye[17] = C*1.7701307697799304f*y*z*(3.f*x2 - y2);
  Ye[18] = C*0.9461746957575601f*x*y*(7.f*z2 - 1.f);
  Ye[19] = C*0.6690465435572892f*y*z*(7.f*z2 - 3.f);
  Ye[20] = C*0.10578554691520431f*(35.f*z2*z2 - 30.f*z2 + 3.f);
  Ye[21] = C*0.6690465435572892f*x*z*(7.f*z2 - 3.f);
  Ye[22] = C*0.47308734787878004f*(x2 - y2)*(7.f*z2 - 1.f);
  Ye[23] = C*1.7701307697799304f*x*z*(x2 - 3.f*y2);
  Ye[24] = C*0.6258357354491761f*(x2*x2 - 6.f*x2*y2 + y2*y2);
}

__global__ void rbf_kernel(const float* __restrict__ rbuf, float* __restrict__ rbf)
{
  int t = blockIdx.x * blockDim.x + threadIdx.x;
  if (t >= EDG * 32) return;
  int e = t >> 5, k = t & 31;
  float x = __expf(-0.5f * rbuf[e]);
  x = fminf(fmaxf(x, 1e-6f), 1.f - 1e-6f);
  float kf = (float)k;
  float logc = lgammaf(32.f) - lgammaf(kf + 1.f) - lgammaf(32.f - kf);
  rbf[t] = __expf(logc + kf * __logf(x) + (31.f - kf) * log1pf(-x));
}

__global__ void init_nodes_kernel(const int* __restrict__ z, const float* __restrict__ embed,
                                  float* __restrict__ nscal, float* __restrict__ xs)
{
  int t = blockIdx.x * blockDim.x + threadIdx.x;           // [n][s][c]
  if (t >= NND * SDIM * HSZ) return;
  int c = t & 127;
  int s = (t >> 7) % SDIM;
  int n = t / (SDIM * HSZ);
  float v = 0.f;
  if (s == 0) { v = embed[(long)z[n] * HSZ + c]; nscal[n * HSZ + c] = v; }
  xs[t] = v;
}

// msg[e,s,c] = w[e,l,0,c]*xs[src][s][c] + w[e,l,1,c]*xs[src][0][c]*Y[e][s]
__global__ void msg_kernel(const float* __restrict__ w, const float* __restrict__ xs,
                           const float* __restrict__ Y, const int* __restrict__ eidx,
                           float* __restrict__ msg)
{
  int t = blockIdx.x * blockDim.x + threadIdx.x;
  if (t >= EDG * SDIM * HSZ) return;
  int c = t & 127;
  int s = (t >> 7) % SDIM;
  int e = t / (SDIM * HSZ);
  int l = c_SL[s];
  int src = eidx[EDG + e];
  const float* wl = w + (long)e * RADW + l * 256;
  float xv = xs[((long)src * SDIM + s) * HSZ + c];
  float x0 = xs[(long)src * SDIM * HSZ + c];
  msg[t] = wl[c] * xv + wl[128 + c] * x0 * Y[(long)e * SDIM + s];
}

// edges sorted by dst: node n owns edges [n*23, (n+1)*23)
__global__ void agg_kernel(const float* __restrict__ msg, float* __restrict__ agg)
{
  int t = blockIdx.x * blockDim.x + threadIdx.x;
  if (t >= NND * SDIM * HSZ) return;
  int n  = t / (SDIM * HSZ);
  int sc = t % (SDIM * HSZ);
  const float* mp = msg + (long)n * NPE * SDIM * HSZ + sc;
  float a = 0.f;
#pragma unroll 4
  for (int j = 0; j < NPE; ++j) a += mp[(long)j * SDIM * HSZ];
  agg[t] = a * 0.20851441405707477f;   // 1/sqrt(23)
}

// g[e,s,c] = wg[e,l,0,c]*xs[dst][s][c] + wg[e,l,1,c]*xs[src][s][c]
__global__ void pair_gather_kernel(const float* __restrict__ wg, const float* __restrict__ xs,
                                   const int* __restrict__ eidx, float* __restrict__ g)
{
  int t = blockIdx.x * blockDim.x + threadIdx.x;
  if (t >= EDG * SDIM * HSZ) return;
  int c = t & 127;
  int s = (t >> 7) % SDIM;
  int e = t / (SDIM * HSZ);
  int l = c_SL[s];
  int d = eidx[e], sr = eidx[EDG + e];
  const float* wl = wg + (long)e * RADW + l * 256;
  g[t] = wl[c]       * xs[((long)d  * SDIM + s) * HSZ + c]
       + wl[128 + c] * xs[((long)sr * SDIM + s) * HSZ + c];
}

__global__ void concat_kernel(const float* __restrict__ nscal, const int* __restrict__ eidx,
                              float* __restrict__ psc)
{
  int t = blockIdx.x * blockDim.x + threadIdx.x;
  if (t >= EDG * 2 * HSZ) return;
  int c = t & 255, e = t >> 8;
  int node = (c < HSZ) ? eidx[e] : eidx[EDG + e];
  psc[t] = nscal[(long)node * HSZ + (c & 127)];
}

// tensor-product expansion into ORBxORB block (one thread per node/edge)
__global__ void expansion_kernel(const float* __restrict__ fb,  // [B][25][32]
                                 const float* __restrict__ pw,  // [B][56][32]
                                 const float* __restrict__ pb,  // [B][14]
                                 const float* __restrict__ Tp,  // packed CG tensors
                                 float* __restrict__ H, int B)
{
  int b = blockIdx.x * blockDim.x + threadIdx.x;
  if (b >= B) return;
  float h[ORB][ORB];
  for (int i = 0; i < ORB; ++i)
    for (int j = 0; j < ORB; ++j) h[i][j] = 0.f;
  for (int t = 0; t < NTERM; ++t) {
    int l = c_TL[t], l1 = c_TL1[t], l2 = c_TL2[t];
    int m1 = c_MUL[l1], m2 = c_MUL[l2];
    int d1 = 2*l1+1, d2 = 2*l2+1, dl = 2*l+1;
    const float* Tt = Tp + c_TOFF[t];
    int r0 = c_ROFF[l1], c0 = c_ROFF[l2];
    for (int a = 0; a < m1; ++a)
      for (int bb = 0; bb < m2; ++bb) {
        const float* wv = pw + ((long)b * NPWC + c_TPW[t] + a * m2 + bb) * HBSZ;
        float s1[9];
        for (int m = 0; m < dl; ++m) {
          const float* fp = fb + ((long)b * SDIM + c_LOFF[l] + m) * HBSZ;
          float acc = 0.f;
          for (int c = 0; c < HBSZ; ++c) acc += fp[c] * wv[c];
          s1[m] = acc;
        }
        for (int d = 0; d < d1; ++d)
          for (int e2 = 0; e2 < d2; ++e2) {
            float acc = 0.f;
            for (int m = 0; m < dl; ++m) acc += s1[m] * Tt[(m * d1 + d) * d2 + e2];
            h[r0 + a * d1 + d][c0 + bb * d2 + e2] += acc;
          }
      }
  }
  const float* pp = pb + (long)b * ORB;
  for (int q = 0; q < 3; ++q) for (int r = 0; r < 3; ++r) h[q][r] += pp[q*3 + r];
  for (int q = 0; q < 2; ++q) for (int r = 0; r < 2; ++r)
    for (int t3 = 0; t3 < 3; ++t3) h[3 + q*3 + t3][3 + r*3 + t3] += pp[9 + q*2 + r];
  for (int t5 = 0; t5 < 5; ++t5) h[9 + t5][9 + t5] += pp[13];
  float* out = H + (long)b * ORB * ORB;
  for (int i = 0; i < ORB; ++i)
    for (int j = 0; j < ORB; ++j) out[i * ORB + j] = h[i][j];
}

__global__ void sym_diag_kernel(const float* __restrict__ Hii, float* __restrict__ out)
{
  int t = blockIdx.x * blockDim.x + threadIdx.x;
  if (t >= NND * ORB * ORB) return;
  int n = t / (ORB * ORB), r = t % (ORB * ORB);
  int i = r / ORB, j = r % ORB;
  out[t] = Hii[(long)n * 196 + i * ORB + j] + Hii[(long)n * 196 + j * ORB + i];
}

__global__ void sym_off_kernel(const float* __restrict__ Hij, const int* __restrict__ tr,
                               float* __restrict__ out)
{
  int t = blockIdx.x * blockDim.x + threadIdx.x;
  if (t >= EDG * ORB * ORB) return;
  int e = t / (ORB * ORB), r = t % (ORB * ORB);
  int i = r / ORB, j = r % ORB;
  long te = tr[e];
  out[t] = Hij[(long)e * 196 + i * ORB + j] + Hij[te * 196 + j * ORB + i];
}

// ------------------------------- host side ------------------------------------
static inline unsigned div_magic(int d) {   // exact floor(x/d) via (x*m)>>40, d<=9
  return (unsigned)((((unsigned long long)1 << 40) + (unsigned long long)d - 1) /
                    (unsigned long long)d);
}
static inline void launch_gemm(hipStream_t st,
                               const float* A, int ai, long ao, int al,
                               const float* B, int ldb, const float* bias,
                               float* C, int ci, long co, int cl,
                               int M, int Nn, int K, int flags)
{
  dim3 g((Nn + 63) / 64, (M + 63) / 64), b(128);
  gemm_f16_wmma<<<g, b, 0, st>>>(A, ai, div_magic(ai), ao, al, B, ldb, bias,
                                 C, ci, div_magic(ci), co, cl, M, Nn, K, flags);
}

extern "C" void kernel_launch(void* const* d_in, const int* in_sizes, int n_in,
                              void* d_out, int out_size, void* d_ws, size_t ws_size,
                              hipStream_t stream)
{
  (void)in_sizes; (void)n_in; (void)out_size; (void)ws_size;
  // ---- input decode: top-level dict insertion order; params jax-tree sorted ----
  int gi = 0;
  const int*   z     = (const int*)  d_in[gi++];
  const float* pos   = (const float*)d_in[gi++];
  const int*   eidx  = (const int*)  d_in[gi++];   // [2][E] (dst row, src row)
  const int*   tridx = (const int*)  d_in[gi++];
  // params: 'T'(19, name-sorted) < 'conv' < 'embed' < 'fc_ii' < 'fc_ii_b'
  //          < 'fc_ij' < 'fc_ij_b' < 'out_ii' < 'out_ij' < 'pair' < 'selfl'
  const float* Tm[19];
  for (int i = 0; i < 19; ++i) Tm[i] = (const float*)d_in[gi++];
  struct Mlp { const float *b1, *b2, *w1, *w2; };
  struct Cv  { const float* mix; Mlp rad; };
  Cv conv[5];
  for (int i = 0; i < 5; ++i) {                   // {'mix','rad':{b1,b2,w1,w2}}
    conv[i].mix    = (const float*)d_in[gi++];
    conv[i].rad.b1 = (const float*)d_in[gi++];
    conv[i].rad.b2 = (const float*)d_in[gi++];
    conv[i].rad.w1 = (const float*)d_in[gi++];
    conv[i].rad.w2 = (const float*)d_in[gi++];
  }
  const float* embed = (const float*)d_in[gi++];
  Mlp fcs[4];                                     // fc_ii, fc_ii_b, fc_ij, fc_ij_b
  for (int i = 0; i < 4; ++i) {
    fcs[i].b1 = (const float*)d_in[gi++];
    fcs[i].b2 = (const float*)d_in[gi++];
    fcs[i].w1 = (const float*)d_in[gi++];
    fcs[i].w2 = (const float*)d_in[gi++];
  }
  const float* out_ii = (const float*)d_in[gi++]; // [5][128][32]
  const float* out_ij = (const float*)d_in[gi++];
  Cv pairp[2];
  for (int i = 0; i < 2; ++i) {
    pairp[i].mix    = (const float*)d_in[gi++];
    pairp[i].rad.b1 = (const float*)d_in[gi++];
    pairp[i].rad.b2 = (const float*)d_in[gi++];
    pairp[i].rad.w1 = (const float*)d_in[gi++];
    pairp[i].rad.w2 = (const float*)d_in[gi++];
  }
  const float* selfw[2];
  for (int i = 0; i < 2; ++i) selfw[i] = (const float*)d_in[gi++];

  // ---- workspace layout (floats) ----
  float* W = (float*)d_ws;
  size_t off = 0;
  auto alloc = [&](size_t n) { float* p = W + off; off += n; return p; };
  float* Yb   = alloc((size_t)EDG * SDIM);
  float* rb   = alloc((size_t)EDG);
  float* rbfb = alloc((size_t)EDG * 32);
  float* nsc  = alloc((size_t)NND * HSZ);
  float* xs   = alloc((size_t)NND * SDIM * HSZ);
  float* agg  = alloc((size_t)NND * SDIM * HSZ);
  float* fii  = alloc((size_t)NND * SDIM * HSZ);
  float* hbuf = alloc((size_t)EDG * HSZ);
  float* wbuf = alloc((size_t)EDG * RADW);
  float* gbuf = alloc((size_t)EDG * SDIM * HSZ);   // msg, reused as pair-gather
  float* fij  = alloc((size_t)EDG * SDIM * HSZ);
  float* fiib = alloc((size_t)NND * SDIM * HBSZ);
  float* fijb = alloc((size_t)EDG * SDIM * HBSZ);
  float* pwii = alloc((size_t)NND * NPWC * HBSZ);
  float* pbii = alloc((size_t)NND * ORB);
  float* psc  = alloc((size_t)EDG * 2 * HSZ);
  float* pwij = alloc((size_t)EDG * NPWC * HBSZ);
  float* pbij = alloc((size_t)EDG * ORB);
  float* Hii  = alloc((size_t)NND * ORB * ORB);
  float* Hij  = alloc((size_t)EDG * ORB * ORB);
  float* Tpk  = alloc(1225);

  // pack the 19 CG tensors (alphabetical input slots -> PATHS visit order)
  static const int vsz[19]    = {1,9,25,9,9,27,45,45,75,105,25,45,75,105,25,75,125,175,225};
  static const int valpha[19] = {0,3,9,4,1,5,10,6,11,15,12,7,13,16,2,8,14,17,18};
  size_t to = 0;
  for (int v = 0; v < 19; ++v) {
    (void)hipMemcpyAsync(Tpk + to, Tm[valpha[v]], (size_t)vsz[v] * sizeof(float),
                         hipMemcpyDeviceToDevice, stream);
    to += vsz[v];
  }

  geom_kernel<<<(EDG + 255) / 256, 256, 0, stream>>>(pos, eidx, Yb, rb);
  rbf_kernel<<<(EDG * 32 + 255) / 256, 256, 0, stream>>>(rb, rbfb);
  init_nodes_kernel<<<(NND * SDIM * HSZ + 255) / 256, 256, 0, stream>>>(z, embed, nsc, xs);

  static const int DL[5]   = {1, 3, 5, 7, 9};
  static const int LOFF[5] = {0, 1, 4, 9, 16};

  for (int i = 0; i < 5; ++i) {
    // radial MLP: rbf -> silu -> [E,1280]
    launch_gemm(stream, rbfb, 1, 32, 0, conv[i].rad.w1, HSZ, conv[i].rad.b1,
                hbuf, 1, HSZ, 0, EDG, HSZ, 32, 1);
    launch_gemm(stream, hbuf, 1, HSZ, 0, conv[i].rad.w2, RADW, conv[i].rad.b2,
                wbuf, 1, RADW, 0, EDG, RADW, HSZ, 0);
    msg_kernel<<<(EDG * SDIM * HSZ + 255) / 256, 256, 0, stream>>>(wbuf, xs, Yb, eidx, gbuf);
    agg_kernel<<<(NND * SDIM * HSZ + 255) / 256, 256, 0, stream>>>(gbuf, agg);
    for (int l = 0; l < 5; ++l)   // xs[l] = (i>0 ? xs[l] : 0) + agg[l] @ mix[l]
      launch_gemm(stream, agg + LOFF[l] * HSZ, DL[l], SDIM * HSZ, HSZ,
                  conv[i].mix + l * HSZ * HSZ, HSZ, nullptr,
                  xs + LOFF[l] * HSZ, DL[l], SDIM * HSZ, HSZ,
                  NND * DL[l], HSZ, HSZ, (i > 0) ? 2 : 0);
    if (i > 2) {
      int si = i - 3;
      for (int l = 0; l < 5; ++l)  // fii += xs[l] @ selfl.w[l]
        launch_gemm(stream, xs + LOFF[l] * HSZ, DL[l], SDIM * HSZ, HSZ,
                    selfw[si] + l * HSZ * HSZ, HSZ, nullptr,
                    fii + LOFF[l] * HSZ, DL[l], SDIM * HSZ, HSZ,
                    NND * DL[l], HSZ, HSZ, si ? 2 : 0);
      launch_gemm(stream, rbfb, 1, 32, 0, pairp[si].rad.w1, HSZ, pairp[si].rad.b1,
                  hbuf, 1, HSZ, 0, EDG, HSZ, 32, 1);
      launch_gemm(stream, hbuf, 1, HSZ, 0, pairp[si].rad.w2, RADW, pairp[si].rad.b2,
                  wbuf, 1, RADW, 0, EDG, RADW, HSZ, 0);
      pair_gather_kernel<<<(EDG * SDIM * HSZ + 255) / 256, 256, 0, stream>>>(wbuf, xs, eidx, gbuf);
      for (int l = 0; l < 5; ++l)  // fij += g[l] @ pair.mix[l]
        launch_gemm(stream, gbuf + LOFF[l] * HSZ, DL[l], SDIM * HSZ, HSZ,
                    pairp[si].mix + l * HSZ * HSZ, HSZ, nullptr,
                    fij + LOFF[l] * HSZ, DL[l], SDIM * HSZ, HSZ,
                    EDG * DL[l], HSZ, HSZ, si ? 2 : 0);
    }
  }

  for (int l = 0; l < 5; ++l) {    // project to HBS=32
    launch_gemm(stream, fii + LOFF[l] * HSZ, DL[l], SDIM * HSZ, HSZ,
                out_ii + l * HSZ * HBSZ, HBSZ, nullptr,
                fiib + LOFF[l] * HBSZ, DL[l], SDIM * HBSZ, HBSZ,
                NND * DL[l], HBSZ, HSZ, 0);
    launch_gemm(stream, fij + LOFF[l] * HSZ, DL[l], SDIM * HSZ, HSZ,
                out_ij + l * HSZ * HBSZ, HBSZ, nullptr,
                fijb + LOFF[l] * HBSZ, DL[l], SDIM * HBSZ, HBSZ,
                EDG * DL[l], HBSZ, HSZ, 0);
  }
  // fc_ii / fc_ii_b
  launch_gemm(stream, nsc, 1, HSZ, 0, fcs[0].w1, HSZ, fcs[0].b1, hbuf, 1, HSZ, 0, NND, HSZ, HSZ, 1);
  launch_gemm(stream, hbuf, 1, HSZ, 0, fcs[0].w2, NPWC * HBSZ, fcs[0].b2,
              pwii, 1, NPWC * HBSZ, 0, NND, NPWC * HBSZ, HSZ, 0);
  launch_gemm(stream, nsc, 1, HSZ, 0, fcs[1].w1, HSZ, fcs[1].b1, hbuf, 1, HSZ, 0, NND, HSZ, HSZ, 1);
  launch_gemm(stream, hbuf, 1, HSZ, 0, fcs[1].w2, ORB, fcs[1].b2, pbii, 1, ORB, 0, NND, ORB, HSZ, 0);
  // fc_ij / fc_ij_b on concatenated pair features
  concat_kernel<<<(EDG * 2 * HSZ + 255) / 256, 256, 0, stream>>>(nsc, eidx, psc);
  launch_gemm(stream, psc, 1, 2 * HSZ, 0, fcs[2].w1, HSZ, fcs[2].b1, hbuf, 1, HSZ, 0, EDG, HSZ, 2 * HSZ, 1);
  launch_gemm(stream, hbuf, 1, HSZ, 0, fcs[2].w2, NPWC * HBSZ, fcs[2].b2,
              pwij, 1, NPWC * HBSZ, 0, EDG, NPWC * HBSZ, HSZ, 0);
  launch_gemm(stream, psc, 1, 2 * HSZ, 0, fcs[3].w1, HSZ, fcs[3].b1, hbuf, 1, HSZ, 0, EDG, HSZ, 2 * HSZ, 1);
  launch_gemm(stream, hbuf, 1, HSZ, 0, fcs[3].w2, ORB, fcs[3].b2, pbij, 1, ORB, 0, EDG, ORB, HSZ, 0);

  expansion_kernel<<<(NND + 127) / 128, 128, 0, stream>>>(fiib, pwii, pbii, Tpk, Hii, NND);
  expansion_kernel<<<(EDG + 127) / 128, 128, 0, stream>>>(fijb, pwij, pbij, Tpk, Hij, EDG);

  float* outp = (float*)d_out;
  sym_diag_kernel<<<(NND * 196 + 255) / 256, 256, 0, stream>>>(Hii, outp);
  sym_off_kernel<<<(EDG * 196 + 255) / 256, 256, 0, stream>>>(Hij, tridx, outp + (size_t)NND * 196);
}
